// FusedMHA_75067438400028
// MI455X (gfx1250) — matrix operand order repaired and checked by reading, not compiled, not run on hardware
//
#include <hip/hip_runtime.h>
#include <hip/hip_bf16.h>

typedef unsigned short u16;
typedef unsigned int   u32;
typedef unsigned long long u64;
typedef __attribute__((ext_vector_type(16))) __bf16 v16bf;
typedef __attribute__((ext_vector_type(8)))  float  v8f;

union FragU { v16bf v; u32 u[8]; uint4 q[2]; };

__device__ __forceinline__ u16 f2bf(float x) {
    u32 u = __float_as_uint(x);
    u32 r = (u + 0x7FFFu + ((u >> 16) & 1u)) >> 16;
    return (u16)r;
}

// CDNA5 async copy: global -> LDS, 16B per lane, tracked by ASYNCcnt.
__device__ __forceinline__ void async_load_b128(const u16* lds_dst, const u16* gsrc) {
    u32 lds_addr = (u32)(size_t)lds_dst;
    asm volatile("global_load_async_to_lds_b128 %0, %1, off"
                 :: "v"(lds_addr), "v"((u64)gsrc) : "memory");
}
__device__ __forceinline__ void wait_async0() {
    asm volatile("s_wait_asynccnt 0x0" ::: "memory");
}

// A-matrix 16x32 bf16 fragment (ISA 7.12.2): lane<16 -> row=lane, K {0..7,16..23};
// lane>=16 -> row=lane-16, K {8..15,24..31}. p points at tile [0][0], ld in u16 elems.
__device__ __forceinline__ v16bf load_frag_a(const u16* p, int ld, int lane) {
    const u16* q = p + (size_t)(lane & 15) * ld + ((lane & 16) ? 8 : 0);
    FragU f;
    f.q[0] = *(const uint4*)(q);
    f.q[1] = *(const uint4*)(q + 16);
    return f.v;
}

// B-matrix 32x16 bf16 fragment: lane<16 -> col=lane, K {0..7,16..23}; lane>=16 -> col=lane-16, K {8..15,24..31}.
__device__ __forceinline__ v16bf load_frag_b(const u16* p, int ld, int lane) {
    const u16* q = p + (size_t)(lane & 15) * ld + ((lane & 16) ? 8 : 0);
    FragU f;
    f.q[0] = *(const uint4*)(q);
    f.q[1] = *(const uint4*)(q + 16);
    return f.v;
}

__device__ __forceinline__ v8f wmma_bf16(v16bf a, v16bf b, v8f c) {
    return __builtin_amdgcn_wmma_f32_16x16x32_bf16(false, a, false, b, (short)0, c, false, false);
}

__global__ void convert_f32_bf16(const float* __restrict__ in, u16* __restrict__ out, int n) {
    int i = blockIdx.x * blockDim.x + threadIdx.x;
    int stride = gridDim.x * blockDim.x;
    for (; i < n; i += stride) out[i] = f2bf(in[i]);
}

// C[M,N] = A[M,K] * B[N,K]^T, bf16 inputs, f32 accumulate.
// 256 threads = 8 waves; tile 128x128, BK=32, double-buffered async LDS staging.
template <bool OUT_BF16>
__global__ __launch_bounds__(256)
void gemm_bf16_nt(const u16* __restrict__ A, const u16* __restrict__ B,
                  void* __restrict__ Cout, int M, int N, int K) {
    __shared__ __align__(16) u16 As[2][128 * 32];
    __shared__ __align__(16) u16 Bs[2][128 * 32];

    const int tid  = threadIdx.x;
    const int lane = tid & 31;
    const int wave = tid >> 5;
    const int wm   = wave >> 1;   // 0..3 -> 32 rows
    const int wn   = wave & 1;    // 0..1 -> 64 cols
    const int bM   = blockIdx.y * 128;
    const int bN   = blockIdx.x * 128;

    auto stage = [&](int buf, int k0) {
#pragma unroll
        for (int it = 0; it < 2; ++it) {
            int idx = tid + it * 256;
            int row = idx >> 2, seg = idx & 3;
            async_load_b128(&As[buf][row * 32 + seg * 8],
                            A + (size_t)(bM + row) * K + k0 + seg * 8);
            async_load_b128(&Bs[buf][row * 32 + seg * 8],
                            B + (size_t)(bN + row) * K + k0 + seg * 8);
        }
    };

    v8f acc[2][4];
#pragma unroll
    for (int i = 0; i < 2; ++i)
#pragma unroll
        for (int j = 0; j < 4; ++j) acc[i][j] = v8f{};

    const int nk = K >> 5;
    stage(0, 0);
    wait_async0();
    __syncthreads();

    for (int kt = 0; kt < nk; ++kt) {
        const int buf = kt & 1;
        if (kt + 1 < nk) stage(buf ^ 1, (kt + 1) * 32);   // prefetch while computing

        v16bf af[2], bfr[4];
#pragma unroll
        for (int rt = 0; rt < 2; ++rt)
            af[rt] = load_frag_a(&As[buf][(wm * 32 + rt * 16) * 32], 32, lane);
#pragma unroll
        for (int ft = 0; ft < 4; ++ft)
            bfr[ft] = load_frag_b(&Bs[buf][(wn * 64 + ft * 16) * 32], 32, lane);
#pragma unroll
        for (int rt = 0; rt < 2; ++rt)
#pragma unroll
            for (int ft = 0; ft < 4; ++ft)
                acc[rt][ft] = wmma_bf16(af[rt], bfr[ft], acc[rt][ft]);

        wait_async0();
        __syncthreads();
    }

    // C layout: VGPR r -> row r (lanes 0-15) / r+8 (lanes 16-31); col = lane&15.
    const int cn   = lane & 15;
    const int roff = (lane & 16) ? 8 : 0;
#pragma unroll
    for (int rt = 0; rt < 2; ++rt)
#pragma unroll
        for (int ft = 0; ft < 4; ++ft)
#pragma unroll
            for (int r = 0; r < 8; ++r) {
                size_t row = bM + wm * 32 + rt * 16 + roff + r;
                size_t col = bN + wn * 64 + ft * 16 + cn;
                if (OUT_BF16)
                    ((u16*)Cout)[row * N + col] = f2bf(acc[rt][ft][r]);
                else
                    ((float*)Cout)[row * N + col] = acc[rt][ft][r];
            }
}

// Flash-style causal attention over bf16 qkv [8192, 6144] -> y bf16 [8192, 2048].
__global__ __launch_bounds__(256)
void attn_causal(const u16* __restrict__ qkv, u16* __restrict__ y) {
    const int T = 2048, C = 2048, F = 6144, D = 128;
    const int b = blockIdx.x >> 4, h = blockIdx.x & 15;
    const int r0 = blockIdx.y * 128;

    const int tid  = threadIdx.x;
    const int lane = tid & 31;
    const int wave = tid >> 5;
    const int qrow = r0 + wave * 16;           // this wave's 16 query rows

    __shared__ __align__(16) u16 Kl[32 * 128];   // key tile, [key][d]
    __shared__ __align__(16) u16 Vt[128 * 32];   // value tile transposed, [d][key]
    __shared__ __align__(16) u16 Pl[8][16 * 32]; // per-wave P scratch

    // Q fragments for this wave: 4 x (16x32) over D=128.
    const u16* qbase = qkv + ((size_t)(b * T + qrow)) * F + h * D;
    v16bf qf[4];
#pragma unroll
    for (int ks = 0; ks < 4; ++ks) qf[ks] = load_frag_a(qbase + ks * 32, F, lane);

    v8f o[8];
#pragma unroll
    for (int i = 0; i < 8; ++i) o[i] = v8f{};
    float m[8], l[8];
#pragma unroll
    for (int r = 0; r < 8; ++r) { m[r] = -1e30f; l[r] = 0.0f; }

    const float scale = 0.08838834764831845f;  // 1/sqrt(128)
    const int cn   = lane & 15;
    const int roff = (lane & 16) ? 8 : 0;
    const int ntiles = (r0 + 128) >> 5;

    for (int jt = 0; jt < ntiles; ++jt) {
        const int jbase = jt * 32;
        // Stage K tile [32][128] via CDNA5 async copy (2 x b128 per thread).
#pragma unroll
        for (int it = 0; it < 2; ++it) {
            int idx = tid + it * 256;
            int row = idx >> 4, seg = idx & 15;
            async_load_b128(&Kl[row * 128 + seg * 8],
                            qkv + ((size_t)(b * T + jbase + row)) * F + C + h * D + seg * 8);
        }
        // Stage V tile transposed: [d][key] (register transpose).
#pragma unroll
        for (int it = 0; it < 2; ++it) {
            int idx = tid + it * 256;
            int row = idx >> 4, seg = idx & 15;
            uint4 vv = *(const uint4*)(qkv + ((size_t)(b * T + jbase + row)) * F + 2 * C + h * D + seg * 8);
            u16 tmp[8];
            *(uint4*)tmp = vv;
#pragma unroll
            for (int e = 0; e < 8; ++e) Vt[(seg * 8 + e) * 32 + row] = tmp[e];
        }
        wait_async0();
        __syncthreads();

        if (jbase <= qrow + 15) {   // tile has at least one unmasked key for this wave
            // S = Q K^T (16 rows x 32 keys) : two 16x16 C tiles, K-dim = 128.
            v8f s0 = v8f{}, s1 = v8f{};
#pragma unroll
            for (int ks = 0; ks < 4; ++ks) {
                v16bf b0 = load_frag_b(&Kl[0 * 128 + ks * 32], 128, lane);
                v16bf b1 = load_frag_b(&Kl[16 * 128 + ks * 32], 128, lane);
                s0 = wmma_bf16(qf[ks], b0, s0);
                s1 = wmma_bf16(qf[ks], b1, s1);
            }
            // scale + causal mask + online softmax
            float mnew[8];
#pragma unroll
            for (int r = 0; r < 8; ++r) {
                int row = qrow + roff + r;
                float a0 = s0[r] * scale; if (jbase + cn > row)      a0 = -1e30f;
                float a1 = s1[r] * scale; if (jbase + 16 + cn > row) a1 = -1e30f;
                s0[r] = a0; s1[r] = a1;
                mnew[r] = fmaxf(m[r], fmaxf(a0, a1));
            }
#pragma unroll
            for (int off = 1; off < 16; off <<= 1)
#pragma unroll
                for (int r = 0; r < 8; ++r)
                    mnew[r] = fmaxf(mnew[r], __shfl_xor(mnew[r], off, 16));
            float alpha[8], rsum[8];
#pragma unroll
            for (int r = 0; r < 8; ++r) {
                alpha[r] = __expf(m[r] - mnew[r]);
                m[r] = mnew[r];
                float p0 = __expf(s0[r] - mnew[r]);
                float p1 = __expf(s1[r] - mnew[r]);
                s0[r] = p0; s1[r] = p1;
                rsum[r] = p0 + p1;
            }
#pragma unroll
            for (int off = 1; off < 16; off <<= 1)
#pragma unroll
                for (int r = 0; r < 8; ++r)
                    rsum[r] += __shfl_xor(rsum[r], off, 16);
#pragma unroll
            for (int r = 0; r < 8; ++r) l[r] = l[r] * alpha[r] + rsum[r];
#pragma unroll
            for (int nb = 0; nb < 8; ++nb)
#pragma unroll
                for (int r = 0; r < 8; ++r) o[nb][r] *= alpha[r];

            // C-layout -> A-fragment layout via per-wave LDS scratch (bf16).
            u16* pw = &Pl[wave][0];
#pragma unroll
            for (int r = 0; r < 8; ++r) {
                pw[(roff + r) * 32 + cn]      = f2bf(s0[r]);
                pw[(roff + r) * 32 + 16 + cn] = f2bf(s1[r]);
            }
            v16bf pf = load_frag_a(pw, 32, lane);
            // O += P V : 8 N-tiles over D.
#pragma unroll
            for (int nb = 0; nb < 8; ++nb) {
                v16bf vb = load_frag_b(&Vt[(nb * 16) * 32], 32, lane);
                o[nb] = wmma_bf16(pf, vb, o[nb]);
            }
        }
        __syncthreads();
    }

    float inv[8];
#pragma unroll
    for (int r = 0; r < 8; ++r) inv[r] = 1.0f / l[r];
#pragma unroll
    for (int nb = 0; nb < 8; ++nb)
#pragma unroll
        for (int r = 0; r < 8; ++r) {
            size_t row = qrow + roff + r;
            y[(((size_t)b * T + row)) * C + h * D + nb * 16 + cn] = f2bf(o[nb][r] * inv[r]);
        }
}

extern "C" void kernel_launch(void* const* d_in, const int* in_sizes, int n_in,
                              void* d_out, int out_size, void* d_ws, size_t ws_size,
                              hipStream_t stream) {
    const int B = 4, T = 2048, C = 2048, F = 3 * C, M = B * T;
    const float* x     = (const float*)d_in[0];
    const float* W_in  = (const float*)d_in[1];
    const float* W_out = (const float*)d_in[2];
    float* out = (float*)d_out;

    u16* xb   = (u16*)d_ws;                       // [M, C]
    u16* wib  = xb   + (size_t)M * C;             // [F, C]
    u16* wob  = wib  + (size_t)F * C;             // [C, C]
    u16* qkvb = wob  + (size_t)C * C;             // [M, F]
    u16* yb   = qkvb + (size_t)M * F;             // [M, C]

    convert_f32_bf16<<<2048, 256, 0, stream>>>(x,     xb,  M * C);
    convert_f32_bf16<<<2048, 256, 0, stream>>>(W_in,  wib, F * C);
    convert_f32_bf16<<<2048, 256, 0, stream>>>(W_out, wob, C * C);

    // qkv = x @ W_in^T  -> bf16 [M, F]
    gemm_bf16_nt<true><<<dim3(F / 128, M / 128), 256, 0, stream>>>(xb, wib, qkvb, M, F, C);

    // causal attention -> y bf16 [M, C]
    attn_causal<<<dim3(B * 16, T / 128), 256, 0, stream>>>(qkvb, yb);

    // out = y @ W_out^T -> f32 [M, C]
    gemm_bf16_nt<false><<<dim3(C / 128, M / 128), 256, 0, stream>>>(yb, wob, out, M, C, C);
}
